// GraphTransformerFFEncoder_49125835931695
// MI455X (gfx1250) — compile-verified, matrix operand order checked by reading, not run on hardware
//
#include <hip/hip_runtime.h>
#include <hip/hip_bf16.h>
#include <cstdint>
#include <cstddef>

// ---------------------------------------------------------------------------
// GraphTransformerFFEncoder for MI455X (gfx1250, wave32, WMMA)
// GEMMs: bf16 in, f32 accumulate via v_wmma_f32_16x16x32_bf16.
// Register-staged LDS double buffering: global loads for tile t+1 overlap the
// WMMAs of tile t; one barrier per k-step. 128x128 block tile, 8 wave32 waves,
// each wave 32x64 (eight 16x16x32 accumulators).
// ---------------------------------------------------------------------------

#define DEVFN __device__ __forceinline__

typedef __attribute__((ext_vector_type(16))) __bf16 v16bf;
typedef __attribute__((ext_vector_type(8)))  float  v8f;

static constexpr int   kN    = 50000;
static constexpr int   kNpad = 50048;            // 391 * 128
static constexpr int   kE    = 800000;
static constexpr int   kD    = 128;
static constexpr int   kH    = 8;
static constexpr int   kFF   = 512;
static constexpr int   kL    = 3;
static constexpr int   kB    = 50;
static constexpr int   kNPG  = kN / kB;          // 1000 nodes per graph
static constexpr float kEps  = 1e-5f;

// ---------------- helpers ----------------
DEVFN unsigned short f2b(float f) {              // f32 -> bf16, round-nearest-even
  unsigned u = __float_as_uint(f);
  u += 0x7FFFu + ((u >> 16) & 1u);
  return (unsigned short)(u >> 16);
}
DEVFN unsigned f2ord(float f) {                  // order-preserving uint encoding
  unsigned u = __float_as_uint(f);
  return (u & 0x80000000u) ? ~u : (u | 0x80000000u);
}
DEVFN float ord2f(unsigned u) {
  return (u & 0x80000000u) ? __uint_as_float(u & 0x7FFFFFFFu) : __uint_as_float(~u);
}

struct Frag32B { uint4 lo, hi; };
// 16-bit A/B fragment per CDNA5 ISA layout: lane<16 -> K0..7 & K16..23,
// lane>=16 -> K8..15 & K24..31, row = lane%16.
DEVFN v16bf load_frag(const unsigned short* base, int kHalf) {
  Frag32B t;
  t.lo = *reinterpret_cast<const uint4*>(base + kHalf);
  t.hi = *reinterpret_cast<const uint4*>(base + kHalf + 16);
  return __builtin_bit_cast(v16bf, t);
}
DEVFN v8f wmma_bf16(v16bf a, v16bf b, v8f c) {
  return __builtin_amdgcn_wmma_f32_16x16x32_bf16(false, a, false, b, (short)0, c,
                                                 false, false);
}
DEVFN v8f zero8() {
  v8f z;
#pragma unroll
  for (int i = 0; i < 8; ++i) z[i] = 0.f;
  return z;
}

// ---------------- generic zero ----------------
__global__ void zero_f32(float* __restrict__ p, long n) {
  long i = (long)blockIdx.x * blockDim.x + threadIdx.x;
  long stride = (long)gridDim.x * blockDim.x;
  for (; i < n; i += stride) p[i] = 0.f;
}

// ---------------- input / weight casts ----------------
__global__ void cast_x_bf16(const float* __restrict__ x, unsigned short* __restrict__ xb, long n) {
  long i = (long)blockIdx.x * blockDim.x + threadIdx.x;
  if (i < n) xb[i] = f2b(x[i]);
}

// Per layer, produce TRANSPOSED bf16 weights WT[Nc][K]:
//  wqkvsT [512][128]  (cols: Wq|Wk|Wv|Wskip), bias512 f32
//  wf1T   [512][128], bf1 f32
//  wf2T   [128][512], bf2 f32
__global__ void cast_weights(const float* __restrict__ Wq,  const float* __restrict__ bq,
                             const float* __restrict__ Wk,  const float* __restrict__ bk,
                             const float* __restrict__ Wv,  const float* __restrict__ bv,
                             const float* __restrict__ Wsk, const float* __restrict__ bsk,
                             const float* __restrict__ Wf1, const float* __restrict__ bf1,
                             const float* __restrict__ Wf2, const float* __restrict__ bf2,
                             unsigned short* __restrict__ wqkvsT, float* __restrict__ bias512,
                             unsigned short* __restrict__ wf1T,  float* __restrict__ bf1f,
                             unsigned short* __restrict__ wf2T,  float* __restrict__ bf2f) {
  int i = blockIdx.x * blockDim.x + threadIdx.x;
  if (i < 512 * 128) {                      // wqkvsT[n][k] = W(sel)[k][n%128]
    int n = i >> 7, k = i & 127;
    int sel = n >> 7, jj = n & 127;
    const float* W = (sel == 0) ? Wq : (sel == 1) ? Wk : (sel == 2) ? Wv : Wsk;
    wqkvsT[i] = f2b(W[k * 128 + jj]);
    return;
  }
  i -= 512 * 128;
  if (i < 512) {
    int sel = i >> 7, jj = i & 127;
    const float* b = (sel == 0) ? bq : (sel == 1) ? bk : (sel == 2) ? bv : bsk;
    bias512[i] = b[jj];
    return;
  }
  i -= 512;
  if (i < 512 * 128) {                      // wf1T[n][k] = Wf1[k][n]
    int n = i >> 7, k = i & 127;
    wf1T[i] = f2b(Wf1[k * 512 + n]);
    return;
  }
  i -= 512 * 128;
  if (i < 512) { bf1f[i] = bf1[i]; return; }
  i -= 512;
  if (i < 128 * 512) {                      // wf2T[n][k] = Wf2[k][n]
    int n = i >> 9, k = i & 511;
    wf2T[i] = f2b(Wf2[k * 128 + n]);
    return;
  }
  i -= 128 * 512;
  if (i < 128) { bf2f[i] = bf2[i]; }
}

// ---------------- WMMA GEMM ----------------
// C[Mpad,Nc] = A[Mpad,K](bf16) @ WT[Nc][K](bf16, pre-transposed) + bias.
// Block tile 128x128, 8 wave32 waves (4 M-slots x 2 N-slots), wave = 32x64.
// Register-staged double-buffered LDS, one barrier per 32-deep k-step.
template <int K, int LDA, int LDC, int RELU, int OUTBF16>
__global__ __launch_bounds__(256)
void gemm_bf16(const unsigned short* __restrict__ A,
               const unsigned short* __restrict__ WT,
               const float* __restrict__ bias,
               float* __restrict__ Cf, unsigned short* __restrict__ Cb) {
  constexpr int T = K / 32;
  __shared__ unsigned short As[2][128 * 48];   // row stride 48 elems (96B)
  __shared__ unsigned short Bs[2][128 * 48];

  const int tid  = threadIdx.x & 255;          // provably < 256: no guards
  const int lane = tid & 31;
  const int wid  = tid >> 5;
  const int wm   = (wid & 3) * 32;             // wave M offset
  const int wn   = (wid >> 2) * 64;            // wave N offset
  const int m0   = blockIdx.x * 128;
  const int n0   = blockIdx.y * 128;

  // staging geometry: thread covers rows sr and sr+64, 8 elems at col sc
  const int sr = tid >> 2;                     // 0..63
  const int sc = (tid & 3) * 8;                // {0,8,16,24}
  const unsigned short* Ag = A  + (long)(m0 + sr) * LDA + sc;
  const unsigned short* Bg = WT + (long)(n0 + sr) * K   + sc;

  uint4 ra0, ra1, rb0, rb1;
  auto gload = [&](int t) {
    const unsigned short* a = Ag + t * 32;
    const unsigned short* b = Bg + t * 32;
    ra0 = *reinterpret_cast<const uint4*>(a);
    ra1 = *reinterpret_cast<const uint4*>(a + (long)64 * LDA);
    rb0 = *reinterpret_cast<const uint4*>(b);
    rb1 = *reinterpret_cast<const uint4*>(b + (long)64 * K);
  };
  auto lstore = [&](int buf) {
    *reinterpret_cast<uint4*>(&As[buf][sr * 48 + sc])        = ra0;
    *reinterpret_cast<uint4*>(&As[buf][(sr + 64) * 48 + sc]) = ra1;
    *reinterpret_cast<uint4*>(&Bs[buf][sr * 48 + sc])        = rb0;
    *reinterpret_cast<uint4*>(&Bs[buf][(sr + 64) * 48 + sc]) = rb1;
  };

  v8f acc[2][4];
#pragma unroll
  for (int mt = 0; mt < 2; ++mt)
#pragma unroll
    for (int nt = 0; nt < 4; ++nt) acc[mt][nt] = zero8();

  gload(0);
  lstore(0);

  const int kh = (lane < 16) ? 0 : 8;
  const int rA = wm + (lane & 15);
  const int rB = wn + (lane & 15);

#pragma unroll
  for (int t = 0; t < T; ++t) {
    __syncthreads();                           // buf[t&1] stores visible
    if (t + 1 < T) gload(t + 1);               // overlap with compute below
    const int buf = t & 1;
    v16bf a0 = load_frag(&As[buf][rA * 48], kh);
    v16bf a1 = load_frag(&As[buf][(rA + 16) * 48], kh);
    v16bf b0 = load_frag(&Bs[buf][rB * 48], kh);
    v16bf b1 = load_frag(&Bs[buf][(rB + 16) * 48], kh);
    v16bf b2 = load_frag(&Bs[buf][(rB + 32) * 48], kh);
    v16bf b3 = load_frag(&Bs[buf][(rB + 48) * 48], kh);

    acc[0][0] = wmma_bf16(a0, b0, acc[0][0]);
    acc[0][1] = wmma_bf16(a0, b1, acc[0][1]);
    acc[0][2] = wmma_bf16(a0, b2, acc[0][2]);
    acc[0][3] = wmma_bf16(a0, b3, acc[0][3]);
    acc[1][0] = wmma_bf16(a1, b0, acc[1][0]);
    acc[1][1] = wmma_bf16(a1, b1, acc[1][1]);
    acc[1][2] = wmma_bf16(a1, b2, acc[1][2]);
    acc[1][3] = wmma_bf16(a1, b3, acc[1][3]);

    if (t + 1 < T) lstore((t + 1) & 1);        // other buffer: no extra barrier
  }

  // epilogue: element (m,n) lives in VGPR m%8, lane (m/8)*16 + n
#pragma unroll
  for (int mt = 0; mt < 2; ++mt) {
#pragma unroll
    for (int nt = 0; nt < 4; ++nt) {
      int ncol  = n0 + wn + nt * 16 + (lane & 15);
      float bv  = bias[ncol];
      int mbase = m0 + wm + mt * 16 + ((lane >> 4) ? 8 : 0);
#pragma unroll
      for (int r = 0; r < 8; ++r) {
        int row = mbase + r;
        float v = acc[mt][nt][r] + bv;
        if (RELU) v = fmaxf(v, 0.f);
        if (OUTBF16) Cb[(long)row * LDC + ncol] = f2b(v);
        else         Cf[(long)row * LDC + ncol] = v;
      }
    }
  }
}

// ---------------- edge attention ----------------
// qkvs layout per node: [0,128)=Q, [128,256)=K, [256,384)=V, [384,512)=skip.
__global__ __launch_bounds__(256)
void edge_alpha_max(const int* __restrict__ ei, const float* __restrict__ qkvs,
                    float* __restrict__ alphaB, unsigned* __restrict__ amaxU) {
  int tid = blockIdx.x * blockDim.x + threadIdx.x;
  if (tid >= kE * kH) return;
  int e = tid >> 3, h = tid & 7;
  int src = ei[e], dst = ei[kE + e];
  const float4* qp = reinterpret_cast<const float4*>(qkvs + (long)dst * 512 + h * 16);
  const float4* kp = reinterpret_cast<const float4*>(qkvs + (long)src * 512 + 128 + h * 16);
  float s = 0.f;
#pragma unroll
  for (int j = 0; j < 4; ++j) {
    float4 a = qp[j], b = kp[j];
    s += a.x * b.x + a.y * b.y + a.z * b.z + a.w * b.w;
  }
  s *= 0.25f;                                  // 1/sqrt(DH=16)
  alphaB[tid] = s;
  atomicMax(&amaxU[dst * 8 + h], f2ord(s));
}

__global__ __launch_bounds__(256)
void edge_softmax_scatter(const int* __restrict__ ei, const float* __restrict__ qkvs,
                          const float* __restrict__ alphaB, const unsigned* __restrict__ amaxU,
                          float* __restrict__ denom, float* __restrict__ accum) {
  int tid = blockIdx.x * blockDim.x + threadIdx.x;
  if (tid >= kE * kH) return;
  int e = tid >> 3, h = tid & 7;
  int src = ei[e], dst = ei[kE + e];
  float a = alphaB[tid];
  float m = ord2f(amaxU[dst * 8 + h]);
  float w = __expf(a - m);
  atomicAdd(&denom[dst * 8 + h], w);
  const float4* vp = reinterpret_cast<const float4*>(qkvs + (long)src * 512 + 256 + h * 16);
  float* op = accum + (long)dst * 128 + h * 16;
#pragma unroll
  for (int j = 0; j < 4; ++j) {
    float4 v = vp[j];
    atomicAdd(op + j * 4 + 0, w * v.x);
    atomicAdd(op + j * 4 + 1, w * v.y);
    atomicAdd(op + j * 4 + 2, w * v.z);
    atomicAdd(op + j * 4 + 3, w * v.w);
  }
}

// ---------------- node finish (attn/denom + skip) + BN stats ----------------
__global__ __launch_bounds__(256)
void finish_stats(const float* __restrict__ accum, const float* __restrict__ denom,
                  const float* __restrict__ qkvs, float* __restrict__ y,
                  float* __restrict__ stats) {
  __shared__ float ss[128], sq[128];
  int t = threadIdx.x;
  if (t < 128) { ss[t] = 0.f; sq[t] = 0.f; }
  __syncthreads();
  long total  = (long)kN * 128;
  long stride = (long)gridDim.x * blockDim.x;      // multiple of 128 -> channel fixed
  float ls = 0.f, lq = 0.f;
  int c = -1;
  for (long i = (long)blockIdx.x * blockDim.x + t; i < total; i += stride) {
    int n = (int)(i >> 7);
    int cc = (int)(i & 127);
    c = cc;
    float v = accum[i] / (denom[n * 8 + (cc >> 4)] + 1e-16f) + qkvs[(long)n * 512 + 384 + cc];
    y[i] = v;
    ls += v;
    lq += v * v;
  }
  if (c >= 0) { atomicAdd(&ss[c], ls); atomicAdd(&sq[c], lq); }
  __syncthreads();
  if (t < 128) { atomicAdd(&stats[t], ss[t]); atomicAdd(&stats[128 + t], sq[t]); }
}

__global__ __launch_bounds__(256)
void bn_stats(const float* __restrict__ y, float* __restrict__ stats) {
  __shared__ float ss[128], sq[128];
  int t = threadIdx.x;
  if (t < 128) { ss[t] = 0.f; sq[t] = 0.f; }
  __syncthreads();
  long total  = (long)kN * 128;
  long stride = (long)gridDim.x * blockDim.x;
  float ls = 0.f, lq = 0.f;
  int c = -1;
  for (long i = (long)blockIdx.x * blockDim.x + t; i < total; i += stride) {
    c = (int)(i & 127);
    float v = y[i];
    ls += v;
    lq += v * v;
  }
  if (c >= 0) { atomicAdd(&ss[c], ls); atomicAdd(&sq[c], lq); }
  __syncthreads();
  if (t < 128) { atomicAdd(&stats[t], ss[t]); atomicAdd(&stats[128 + t], sq[t]); }
}

template <int WRITE_F32>
__global__ __launch_bounds__(256)
void bn_apply(const float* __restrict__ y, const float* __restrict__ stats,
              const float* __restrict__ g, const float* __restrict__ b,
              float* __restrict__ xf, unsigned short* __restrict__ xb) {
  long i = (long)blockIdx.x * blockDim.x + threadIdx.x;
  if (i >= (long)kN * 128) return;
  int c = (int)(i & 127);
  float mean = stats[c] * (1.f / kN);
  float var  = stats[128 + c] * (1.f / kN) - mean * mean;
  float rstd = rsqrtf(var + kEps);
  float v = (y[i] - mean) * rstd * g[c] + b[c];
  if (WRITE_F32) xf[i] = v;
  xb[i] = f2b(v);
}

// ---------------- graph mean pooling ----------------
__global__ __launch_bounds__(128)
void graph_pool(const float* __restrict__ xf, float* __restrict__ out) {
  int b = blockIdx.x;
  int c = threadIdx.x;
  const float* p = xf + (long)b * kNPG * 128 + c;
  float s = 0.f;
  for (int i = 0; i < kNPG; ++i) s += p[(long)i * 128];
  out[b * 128 + c] = s * (1.f / kNPG);
}

// ---------------------------------------------------------------------------
extern "C" void kernel_launch(void* const* d_in, const int* in_sizes, int n_in,
                              void* d_out, int out_size, void* d_ws, size_t ws_size,
                              hipStream_t stream) {
  (void)in_sizes; (void)n_in; (void)out_size; (void)ws_size;

  const float* x   = (const float*)d_in[0];
  const int*   ei  = (const int*)d_in[1];
  // d_in[2] = batch (unused; graphs are equal-sized and sorted)
  const float* Wq  = (const float*)d_in[3];  const float* bq  = (const float*)d_in[4];
  const float* Wk  = (const float*)d_in[5];  const float* bk  = (const float*)d_in[6];
  const float* Wv  = (const float*)d_in[7];  const float* bv  = (const float*)d_in[8];
  const float* Wsk = (const float*)d_in[9];  const float* bsk = (const float*)d_in[10];
  const float* g1  = (const float*)d_in[11]; const float* be1 = (const float*)d_in[12];
  const float* Wf1 = (const float*)d_in[13]; const float* bf1 = (const float*)d_in[14];
  const float* Wf2 = (const float*)d_in[15]; const float* bf2 = (const float*)d_in[16];
  const float* g2  = (const float*)d_in[17]; const float* be2 = (const float*)d_in[18];
  float* out = (float*)d_out;

  // ---- carve workspace (~273 MB; all offsets 256B aligned) ----
  char*  ws  = (char*)d_ws;
  size_t off = 0;
  auto alloc = [&](size_t bytes) -> void* {
    void* p = ws + off;
    off = (off + bytes + 255) & ~(size_t)255;
    return p;
  };
  unsigned short* xb    = (unsigned short*)alloc((size_t)kNpad * kD * 2);   // bf16 x
  float*          qkvs  = (float*)alloc((size_t)kNpad * 512 * 4);           // Q|K|V|skip
  float*          accum = (float*)alloc((size_t)kNpad * kD * 4);            // attn scatter
  float*          denom = (float*)alloc((size_t)kN * kH * 4);
  unsigned*       amaxU = (unsigned*)alloc((size_t)kN * kH * 4);
  float*          alphaB= (float*)alloc((size_t)kE * kH * 4);
  float*          ybuf  = (float*)alloc((size_t)kNpad * kD * 4);            // pre-BN act
  unsigned short* hb    = (unsigned short*)alloc((size_t)kNpad * kFF * 2);  // FFN hidden
  float*          xf    = (float*)alloc((size_t)kNpad * kD * 4);            // layer out
  float*          stats = (float*)alloc(256 * 4);                           // sum|sumsq
  unsigned short* wqkvsT= (unsigned short*)alloc((size_t)512 * 128 * 2);
  float*          bias512 = (float*)alloc(512 * 4);
  unsigned short* wf1T  = (unsigned short*)alloc((size_t)512 * 128 * 2);
  float*          bf1f  = (float*)alloc(512 * 4);
  unsigned short* wf2T  = (unsigned short*)alloc((size_t)128 * 512 * 2);
  float*          bf2f  = (float*)alloc(128 * 4);

  const int  threads = 256;
  const long nND     = (long)kN * kD;                       // 6.4M
  const int  ndBlk   = (int)((nND + threads - 1) / threads);
  const int  ehBlk   = (kE * kH + threads - 1) / threads;   // 25000
  const int  mTiles  = kNpad / 128;                         // 391

  // input x -> bf16
  cast_x_bf16<<<ndBlk, threads, 0, stream>>>(x, xb, nND);

  for (int l = 0; l < kL; ++l) {
    cast_weights<<<(197760 + threads - 1) / threads, threads, 0, stream>>>(
        Wq + (size_t)l * kD * kD, bq + (size_t)l * kD,
        Wk + (size_t)l * kD * kD, bk + (size_t)l * kD,
        Wv + (size_t)l * kD * kD, bv + (size_t)l * kD,
        Wsk + (size_t)l * kD * kD, bsk + (size_t)l * kD,
        Wf1 + (size_t)l * kD * kFF, bf1 + (size_t)l * kFF,
        Wf2 + (size_t)l * kFF * kD, bf2 + (size_t)l * kD,
        wqkvsT, bias512, wf1T, bf1f, wf2T, bf2f);

    // zero accum + denom + amax (contiguous)
    zero_f32<<<2048, threads, 0, stream>>>((float*)accum,
                                           (long)kNpad * kD + (long)kN * kH * 2);

    // fused Q|K|V|skip GEMM: [Npad,128] x [128,512]
    gemm_bf16<128, 128, 512, 0, 0>
        <<<dim3(mTiles, 512 / 128), threads, 0, stream>>>(xb, wqkvsT, bias512,
                                                          qkvs, nullptr);

    edge_alpha_max<<<ehBlk, threads, 0, stream>>>(ei, qkvs, alphaB, amaxU);
    edge_softmax_scatter<<<ehBlk, threads, 0, stream>>>(ei, qkvs, alphaB, amaxU,
                                                        denom, accum);

    // finish (divide + skip) and BN1
    zero_f32<<<1, threads, 0, stream>>>(stats, 256);
    finish_stats<<<1024, threads, 0, stream>>>(accum, denom, qkvs, ybuf, stats);
    bn_apply<0><<<ndBlk, threads, 0, stream>>>(ybuf, stats,
                                               g1 + (size_t)l * kD,
                                               be1 + (size_t)l * kD, nullptr, xb);

    // FFN1: [Npad,128] x [128,512], ReLU, bf16 out
    gemm_bf16<128, 128, 512, 1, 1>
        <<<dim3(mTiles, kFF / 128), threads, 0, stream>>>(xb, wf1T, bf1f,
                                                          nullptr, hb);

    // FFN2: [Npad,512] x [512,128], f32 out
    gemm_bf16<512, 512, 128, 0, 0>
        <<<dim3(mTiles, kD / 128), threads, 0, stream>>>(hb, wf2T, bf2f,
                                                         ybuf, nullptr);

    // BN2 (writes xf f32 + xb bf16 for next layer)
    zero_f32<<<1, threads, 0, stream>>>(stats, 256);
    bn_stats<<<1024, threads, 0, stream>>>(ybuf, stats);
    bn_apply<1><<<ndBlk, threads, 0, stream>>>(ybuf, stats,
                                               g2 + (size_t)l * kD,
                                               be2 + (size_t)l * kD, xf, xb);
  }

  // dense_embeddings = x reshape -> straight copy
  hipMemcpyAsync(out, xf, (size_t)kN * kD * sizeof(float),
                 hipMemcpyDeviceToDevice, stream);
  // graph_embeddings = per-graph mean
  graph_pool<<<kB, 128, 0, stream>>>(xf, out + (size_t)kN * kD);
}
